// EmbeddingBlock_7816840478890
// MI455X (gfx1250) — matrix operand order repaired and checked by reading, not compiled, not run on hardware
//
#include <hip/hip_runtime.h>
#include <hip/hip_bf16.h>
#include <math.h>

// ---------------------------------------------------------------------------
// Problem constants (match reference)
// ---------------------------------------------------------------------------
#define NH   128      // hidden channels H
#define NR   6        // radial
#define NSH  3        // spin hidden
#define NCODE 285     // 95 atom types * 3 spin classes
#define TILE_E 128    // edges per workgroup
#define LDS_P  136    // padded row stride (halves) for r tile: 272B -> 4-bank rotate

typedef __attribute__((ext_vector_type(16))) _Float16 v16h;
typedef __attribute__((ext_vector_type(8)))  _Float16 v8h;
typedef __attribute__((ext_vector_type(8)))  float    v8f;

// silu: prefer gfx1250 hardware tanh (3 ops, 1 trans); fallback exp+rcp (5 ops).
__device__ __forceinline__ float silu_f(float x) {
#if __has_builtin(__builtin_amdgcn_tanhf)
    const float hx = 0.5f * x;
    return fmaf(hx, __builtin_amdgcn_tanhf(hx), hx);
#elif __has_builtin(__builtin_amdgcn_tanh_f32)
    const float hx = 0.5f * x;
    return fmaf(hx, __builtin_amdgcn_tanh_f32(hx), hx);
#else
    return x * __builtin_amdgcn_rcpf(1.0f + __expf(-x));
#endif
}

// ---------------------------------------------------------------------------
// Kernel 1: build the 285-entry node tables.
//   TA[c] = W1 @ emb[t] + W2 @ spinfeat(sp) + lin_b     (node as source i)
//   TB[c] = W3 @ emb[t] + W4 @ spinfeat(sp)             (node as target j)
// concat column map in lin_w: [0:128)=W1 [128:131)=W2 [131:259)=W3
//                             [259:262)=W4 [262:390)=W5
// ---------------------------------------------------------------------------
__global__ __launch_bounds__(NH)
void ebk_build_tables(const float* __restrict__ emb_w,   // [95,128]
                      const float* __restrict__ spin_w,  // [3,3]
                      const float* __restrict__ spin_b,  // [3]
                      const float* __restrict__ lin_w,   // [128,390]
                      const float* __restrict__ lin_b,   // [128]
                      float* __restrict__ TA,            // [285,128]
                      float* __restrict__ TB)            // [285,128]
{
    const int c  = blockIdx.x;         // 0..284
    const int o  = threadIdx.x;        // 0..127
    const int tp = c / 3;
    const int sp = c % 3;
    const float* er = emb_w + tp * NH;
    const float* lw = lin_w + o * (3 * NH + 2 * NSH);   // row o, 390 wide

    float sa = lin_b[o];
    float sb = 0.0f;
#pragma unroll 8
    for (int d = 0; d < NH; ++d) {
        const float e = er[d];
        sa = fmaf(lw[d], e, sa);           // W1
        sb = fmaf(lw[131 + d], e, sb);     // W3
    }
#pragma unroll
    for (int q = 0; q < NSH; ++q) {
        const float sf = spin_w[q * 3 + sp] + spin_b[q];   // spin_feat[q]
        sa = fmaf(lw[128 + q], sf, sa);    // W2
        sb = fmaf(lw[259 + q], sf, sb);    // W4
    }
    TA[c * NH + o] = sa;
    TB[c * NH + o] = sb;
}

// ---------------------------------------------------------------------------
// Kernel 2: pre-swizzle W5 (= lin_w[:,262:390]) into f16 B-fragment layout.
// Fragment (nt,kc): lane L holds 16 halves h=0..15 with
//   N = nt*16 + (L&15),  K = kc*32 + (L>>4)*16 + h
// stored contiguously: w5h[ (nt*4+kc)*512 + L*16 + h ]
// ---------------------------------------------------------------------------
__global__ __launch_bounds__(256)
void ebk_build_w5h(const float* __restrict__ lin_w,  // [128,390]
                   _Float16* __restrict__ w5h)       // [16384]
{
    const int flat = blockIdx.x * 256 + threadIdx.x;   // 0..16383
    if (flat >= NH * NH) return;
    const int h    = flat & 15;
    const int lane = (flat >> 4) & 31;
    const int frag = flat >> 9;            // nt*4 + kc
    const int kc   = frag & 3;
    const int nt   = frag >> 2;
    const int o    = nt * 16 + (lane & 15);
    const int k    = kc * 32 + ((lane >> 4) << 4) + h;
    w5h[flat] = (_Float16)lin_w[o * (3 * NH + 2 * NSH) + 262 + k];
}

// ---------------------------------------------------------------------------
// Kernel 3: main edge kernel. 256 threads (8 wave32), 128 edges per block.
// FULL=true  : every edge of the tile is in range (no bounds logic at all).
// FULL=false : remainder tile (at most one block), fully guarded.
// ---------------------------------------------------------------------------
template<bool FULL>
__global__ __launch_bounds__(256)
void ebk_edge_kernel(const int*   __restrict__ xtype,  // [N]
                     const int*   __restrict__ sspin,  // [N]
                     const float* __restrict__ rbf,    // [E,6]
                     const int*   __restrict__ eidx_i, // [E]
                     const int*   __restrict__ eidx_j, // [E]
                     const float* __restrict__ rbf_w,  // [128,6]
                     const float* __restrict__ rbf_b,  // [128]
                     const float* __restrict__ TA,     // [285,128]
                     const float* __restrict__ TB,     // [285,128]
                     const _Float16* __restrict__ w5h, // [16384] fragment-swizzled
                     float* __restrict__ out,          // [E,128]
                     int tileBase,                     // first tile index
                     int E)
{
    __shared__ __align__(16) _Float16 lds_r[TILE_E * LDS_P]; // 34816 B
    __shared__ int lds_ci[TILE_E];
    __shared__ int lds_cj[TILE_E];

    const int t    = threadIdx.x;
    const int wave = t >> 5;
    const int lane = t & 31;
    const int e0   = (tileBase + blockIdx.x) * TILE_E;

    // ---- per-edge node codes --------------------------------------------
    if (t < TILE_E) {
        const int e = e0 + t;
        int ci = 0, cj = 0;
        if (FULL || e < E) {
            const int ni = eidx_i[e];
            const int nj = eidx_j[e];
            ci = xtype[ni] * 3 + sspin[ni];
            cj = xtype[nj] * 3 + sspin[nj];
        }
        lds_ci[t] = ci;
        lds_cj[t] = cj;
    }

    // ---- stage 1: r = silu(rbf @ rbf_w.T + rbf_b) -> f16 in LDS ----------
    // Wave w owns the wave-uniform k-block [w*16, w*16+16); lane owns edges
    // {lane, lane+32, lane+64, lane+96}.  k uniformity (via readfirstlane)
    // turns all rbf_w / rbf_b fetches into scalar s_loads -> SGPR operands.
    {
        const int kbase = __builtin_amdgcn_readfirstlane(wave) << 4;
        const float* __restrict__ wk = rbf_w + kbase * NR;   // 16x6 block (SGPR)
        const float* __restrict__ bk = rbf_b + kbase;        // 16 biases  (SGPR)

#pragma unroll
        for (int p = 0; p < 4; ++p) {
            const int eloc = lane + p * 32;
            const int e    = e0 + eloc;
            float rv[NR];
            if (FULL || e < E) {
                const float2 r01 = *(const float2*)(rbf + e * NR);
                const float2 r23 = *(const float2*)(rbf + e * NR + 2);
                const float2 r45 = *(const float2*)(rbf + e * NR + 4);
                rv[0] = r01.x; rv[1] = r01.y;
                rv[2] = r23.x; rv[3] = r23.y;
                rv[4] = r45.x; rv[5] = r45.y;
            } else {
#pragma unroll
                for (int d = 0; d < NR; ++d) rv[d] = 0.0f;
            }

            v16h hv;
#pragma unroll
            for (int kk = 0; kk < 16; ++kk) {
                float a = bk[kk];
#pragma unroll
                for (int d = 0; d < NR; ++d)
                    a = fmaf(rv[d], wk[kk * NR + d], a);
                hv[kk] = (_Float16)silu_f(a);
            }
            // two 16B LDS stores per edge
            _Float16* dst = &lds_r[eloc * LDS_P + kbase];
            *(v8h*)dst = __builtin_shufflevector(hv, hv, 0, 1, 2, 3, 4, 5, 6, 7);
            *(v8h*)(dst + 8) =
                __builtin_shufflevector(hv, hv, 8, 9, 10, 11, 12, 13, 14, 15);
        }
    }
    __syncthreads();

    // ---- stage 2: per-wave WMMA over 8 M-tiles, one N-tile per wave ------
    const int nt = wave;

    // B fragments (K chunks 0..3) live in VGPRs; one 32B contiguous load/lane
    v16h Bfrag[4];
#pragma unroll
    for (int kc = 0; kc < 4; ++kc)
        Bfrag[kc] = *(const v16h*)(w5h + ((nt * 4 + kc) * 32 + lane) * 16);

    const int oc    = nt * 16 + (lane & 15);   // output channel this lane owns
    const int half8 = (lane >> 4) << 3;        // 0 or 8: M-row offset in C
    const int kOff  = (lane >> 4) << 4;        // 0 or 16: K offset in A rows

#pragma unroll 1
    for (int m = 0; m < 8; ++m) {
        // A fragments from LDS (contiguous K, matching w5h B layout):
        // lane<16 : h0..15 -> K+0..15
        // lane>=16: h0..15 -> K+16..31
        const int eRow = m * 16 + (lane & 15);
        const _Float16* rp = &lds_r[eRow * LDS_P];
        v8f acc = {};
#pragma unroll
        for (int kc = 0; kc < 4; ++kc) {
            const v8h a0 = *(const v8h*)(rp + kc * 32 + kOff);
            const v8h a1 = *(const v8h*)(rp + kc * 32 + kOff + 8);
            const v16h A = __builtin_shufflevector(a0, a1,
                0, 1, 2, 3, 4, 5, 6, 7, 8, 9, 10, 11, 12, 13, 14, 15);
            acc = __builtin_amdgcn_wmma_f32_16x16x32_f16(
                false, A, false, Bfrag[kc], (short)0, acc, false, false);
        }

        // ---- epilogue: + TA[code_i] + TB[code_j], silu, store ------------
#pragma unroll
        for (int g = 0; g < 8; ++g) {
            const int eloc = m * 16 + g + half8;   // C row M = g + half8
            const int e    = e0 + eloc;
            const int ci   = lds_ci[eloc];         // LDS broadcast per half-wave
            const int cj   = lds_cj[eloc];
            float v = acc[g] + TA[ci * NH + oc] + TB[cj * NH + oc];
            v = silu_f(v);
            if (FULL || e < E) out[e * NH + oc] = v;
        }
    }
}

// ---------------------------------------------------------------------------
// Host launcher
// ---------------------------------------------------------------------------
extern "C" void kernel_launch(void* const* d_in, const int* in_sizes, int n_in,
                              void* d_out, int out_size, void* d_ws, size_t ws_size,
                              hipStream_t stream) {
    const int*   x      = (const int*)  d_in[0];
    const int*   s      = (const int*)  d_in[1];
    const float* rbf    = (const float*)d_in[2];
    const int*   ei     = (const int*)  d_in[3];
    const int*   ej     = (const int*)  d_in[4];
    const float* emb_w  = (const float*)d_in[5];
    const float* spin_w = (const float*)d_in[6];
    const float* spin_b = (const float*)d_in[7];
    const float* rbf_w  = (const float*)d_in[8];
    const float* rbf_b  = (const float*)d_in[9];
    const float* lin_w  = (const float*)d_in[10];
    const float* lin_b  = (const float*)d_in[11];
    float* out = (float*)d_out;

    const int E = in_sizes[3];

    // workspace: TA (285*128 f32) | TB (285*128 f32) | w5h (128*128 f16) ~ 325 KB
    float*    TA  = (float*)d_ws;
    float*    TB  = TA + NCODE * NH;
    _Float16* w5h = (_Float16*)(TB + NCODE * NH);

    ebk_build_tables<<<NCODE, NH, 0, stream>>>(emb_w, spin_w, spin_b,
                                               lin_w, lin_b, TA, TB);
    ebk_build_w5h<<<(NH * NH + 255) / 256, 256, 0, stream>>>(lin_w, w5h);

    const int nFull = E / TILE_E;               // fully-populated tiles
    const int rem   = E - nFull * TILE_E;       // leftover edges (0..127)

    if (nFull > 0) {
        ebk_edge_kernel<true><<<nFull, 256, 0, stream>>>(
            x, s, rbf, ei, ej, rbf_w, rbf_b, TA, TB, w5h, out, 0, E);
    }
    if (rem > 0) {
        ebk_edge_kernel<false><<<1, 256, 0, stream>>>(
            x, s, rbf, ei, ej, rbf_w, rbf_b, TA, TB, w5h, out, nFull, E);
    }
}
